// LeNet_Binarized_23630910062738
// MI455X (gfx1250) — 8-bit matrix operand order repaired and checked by reading, compiled, not run on hardware
//
#include <hip/hip_runtime.h>

typedef __attribute__((ext_vector_type(8))) int v8i;
typedef __attribute__((ext_vector_type(4))) int v4i;
typedef __attribute__((ext_vector_type(2))) int v2i;

#define EPS 1e-5f

__device__ __forceinline__ float fsign(float x) {
  return (float)((x > 0.0f) - (x < 0.0f));
}
__device__ __forceinline__ signed char csign(float x) {
  return (signed char)((x > 0.0f) - (x < 0.0f));
}
__device__ __forceinline__ int imax2(int a, int b) { return a > b ? a : b; }

__global__ void binpad_kernel(const float* __restrict__ src, signed char* __restrict__ dst,
                              int rsrc, int csrc, int rdst, int cdst) {
  int idx = blockIdx.x * blockDim.x + threadIdx.x;
  int total = rdst * cdst;
  if (idx >= total) return;
  int r = idx / cdst, c = idx % cdst;
  signed char v = 0;
  if (r < rsrc && c < csrc) v = csign(src[(size_t)r * csrc + c]);
  dst[idx] = v;
}

__global__ void conv1_kernel(const float* __restrict__ x, const float* __restrict__ w1,
                             const float* __restrict__ cb1, const float* __restrict__ g1,
                             const float* __restrict__ be1, const float* __restrict__ m1,
                             const float* __restrict__ v1, signed char* __restrict__ a1) {
  int idx = blockIdx.x * blockDim.x + threadIdx.x;
  const int total = 4096 * 18 * 14 * 14;
  if (idx >= total) return;
  int px_ = idx % 14; int t = idx / 14;
  int py_ = t % 14; t /= 14;
  int co = t % 18; int n = t / 18;

  const float* xin = x + (size_t)n * 1024;
  const float* w = w1 + co * 25;
  float ws[25];
#pragma unroll
  for (int i = 0; i < 25; ++i) ws[i] = fsign(w[i]);
  float bias = cb1[co];

  float best = -1e30f;
#pragma unroll
  for (int pw = 0; pw < 4; ++pw) {
    int oy = 2 * py_ + (pw >> 1);
    int ox = 2 * px_ + (pw & 1);
    float acc = bias;
#pragma unroll
    for (int dy = 0; dy < 5; ++dy)
#pragma unroll
      for (int dx = 0; dx < 5; ++dx)
        acc += fsign(xin[(oy + dy) * 32 + (ox + dx)]) * ws[dy * 5 + dx];
    best = fmaxf(best, acc);
  }
  float s = g1[co] * rsqrtf(v1[co] + EPS);
  float val = best * s + (be1[co] - m1[co] * s);
  a1[idx] = csign(val);
}

__global__ void im2col2_kernel(const signed char* __restrict__ a1, signed char* __restrict__ im2) {
  int idx = blockIdx.x * blockDim.x + threadIdx.x;
  const int total = 409600 * 128;
  if (idx >= total) return;
  int m = idx >> 7;
  int k0 = (idx & 127) << 2;
  int n = m / 100, p = m % 100;
  int w = p >> 2, e = p & 3;
  int oy = (w / 5) * 2 + (e >> 1);
  int ox = (w % 5) * 2 + (e & 1);
  const signed char* base = a1 + (size_t)n * 3528;
  unsigned word = 0;
#pragma unroll
  for (int b = 0; b < 4; ++b) {
    int k = k0 + b;
    if (k < 450) {
      int ci = k / 25, r = k % 25;
      int dy = r / 5, dx = r % 5;
      unsigned char val = (unsigned char)base[(ci * 14 + (oy + dy)) * 14 + (ox + dx)];
      word |= ((unsigned)val) << (8 * b);
    }
  }
  *(int*)(im2 + (size_t)m * 512 + k0) = (int)word;
}

__device__ __forceinline__ v8i load_a_frag(const signed char* __restrict__ A, int lda,
                                           int m0, int kb, int Kreal) {
  int lane = threadIdx.x & 31;
  const signed char* row = A + (size_t)(m0 + (lane & 15)) * lda;
  int kbase = kb + ((lane >> 4) << 4);
  v8i a;
#pragma unroll
  for (int q = 0; q < 4; ++q) {
    int k = kbase + ((q & 1) << 3) + ((q >> 1) << 5);
    v2i d = {0, 0};
    if (k + 8 <= Kreal) d = *(const v2i*)(row + k);
    a[2 * q] = d.x;
    a[2 * q + 1] = d.y;
  }
  return a;
}

__device__ __forceinline__ v8i load_b_frag(const signed char* __restrict__ W, int ldw,
                                           int n0, int kb) {
  int lane = threadIdx.x & 31;
  const signed char* row = W + (size_t)(n0 + (lane & 15)) * ldw;
  int kbase = kb + ((lane >> 4) << 4);
  v4i lo = *(const v4i*)(row + kbase);
  v4i hi = *(const v4i*)(row + kbase + 32);
  v8i b;
  b[0] = lo.x; b[1] = lo.y; b[2] = lo.z; b[3] = lo.w;
  b[4] = hi.x; b[5] = hi.y; b[6] = hi.z; b[7] = hi.w;
  return b;
}

__device__ __forceinline__ void conv2_epilogue(v8i c, int nt, int m0,
                                               const float* __restrict__ cb2,
                                               const float* __restrict__ g,
                                               const float* __restrict__ be,
                                               const float* __restrict__ mm,
                                               const float* __restrict__ vv,
                                               signed char* __restrict__ a2) {
  int lane = threadIdx.x & 31;
  int co = nt * 16 + (lane & 15);
  float bias = cb2[co];
  float s = g[co] * rsqrtf(vv[co] + EPS);
  float t0 = be[co] - mm[co] * s;
#pragma unroll
  for (int gq = 0; gq < 2; ++gq) {
    int mrow = m0 + ((lane >> 4) << 3) + gq * 4;
    int i0 = gq * 4;
    int best = imax2(imax2(c[i0], c[i0 + 1]), imax2(c[i0 + 2], c[i0 + 3]));
    float val = ((float)best + bias) * s + t0;
    int n = mrow / 100;
    int w = (mrow % 100) >> 2;
    a2[((size_t)n * 48 + co) * 25 + w] = csign(val);
  }
}

__global__ __launch_bounds__(32) void conv2_wmma_kernel(
    const signed char* __restrict__ im2, const signed char* __restrict__ w2s,
    const float* __restrict__ cb2, const float* __restrict__ g2,
    const float* __restrict__ be2, const float* __restrict__ m2,
    const float* __restrict__ v2, signed char* __restrict__ a2) {
  int m0 = blockIdx.x * 16;
  v8i c0 = {0, 0, 0, 0, 0, 0, 0, 0};
  v8i c1 = {0, 0, 0, 0, 0, 0, 0, 0};
  v8i c2 = {0, 0, 0, 0, 0, 0, 0, 0};
#pragma unroll 1
  for (int kb = 0; kb < 512; kb += 64) {
    v8i a = load_a_frag(im2, 512, m0, kb, 512);
    v8i b0 = load_b_frag(w2s, 512, 0, kb);
    v8i b1 = load_b_frag(w2s, 512, 16, kb);
    v8i b2 = load_b_frag(w2s, 512, 32, kb);
    c0 = __builtin_amdgcn_wmma_i32_16x16x64_iu8(true, a, true, b0, c0, false, false);
    c1 = __builtin_amdgcn_wmma_i32_16x16x64_iu8(true, a, true, b1, c1, false, false);
    c2 = __builtin_amdgcn_wmma_i32_16x16x64_iu8(true, a, true, b2, c2, false, false);
  }
  conv2_epilogue(c0, 0, m0, cb2, g2, be2, m2, v2, a2);
  conv2_epilogue(c1, 1, m0, cb2, g2, be2, m2, v2, a2);
  conv2_epilogue(c2, 2, m0, cb2, g2, be2, m2, v2, a2);
}

__global__ __launch_bounds__(32) void conv3_wmma_kernel(
    const signed char* __restrict__ a2, const signed char* __restrict__ w3s,
    const float* __restrict__ cb3, const float* __restrict__ g, const float* __restrict__ be,
    const float* __restrict__ mm, const float* __restrict__ vv,
    signed char* __restrict__ a3) {
  int mtile = blockIdx.x, ntile = blockIdx.y;
  v8i c = {0, 0, 0, 0, 0, 0, 0, 0};
#pragma unroll 1
  for (int kb = 0; kb < 1216; kb += 64) {
    v8i a = load_a_frag(a2, 1200, mtile * 16, kb, 1200);
    v8i b = load_b_frag(w3s, 1216, ntile * 16, kb);
    c = __builtin_amdgcn_wmma_i32_16x16x64_iu8(true, a, true, b, c, false, false);
  }
  int lane = threadIdx.x & 31;
  int co = ntile * 16 + (lane & 15);
  float bias = 0.f, s = 0.f, t0 = 0.f;
  if (co < 120) {
    bias = cb3[co];
    s = g[co] * rsqrtf(vv[co] + EPS);
    t0 = be[co] - mm[co] * s;
  }
#pragma unroll
  for (int v = 0; v < 8; ++v) {
    int m = mtile * 16 + v + ((lane >> 4) << 3);
    signed char out = 0;
    if (co < 120) out = csign(((float)c[v] + bias) * s + t0);
    a3[(size_t)m * 128 + co] = out;
  }
}

__global__ __launch_bounds__(32) void fc1_wmma_kernel(
    const signed char* __restrict__ a3, const signed char* __restrict__ fw1s,
    const float* __restrict__ fb1, const float* __restrict__ g, const float* __restrict__ be,
    const float* __restrict__ mm, const float* __restrict__ vv,
    signed char* __restrict__ h4) {
  int mtile = blockIdx.x, ntile = blockIdx.y;
  v8i c = {0, 0, 0, 0, 0, 0, 0, 0};
#pragma unroll
  for (int kb = 0; kb < 128; kb += 64) {
    v8i a = load_a_frag(a3, 128, mtile * 16, kb, 128);
    v8i b = load_b_frag(fw1s, 128, ntile * 16, kb);
    c = __builtin_amdgcn_wmma_i32_16x16x64_iu8(true, a, true, b, c, false, false);
  }
  int lane = threadIdx.x & 31;
  int co = ntile * 16 + (lane & 15);
  float bias = 0.f, s = 0.f, t0 = 0.f;
  if (co < 252) {
    bias = fb1[co];
    s = g[co] * rsqrtf(vv[co] + EPS);
    t0 = be[co] - mm[co] * s;
  }
#pragma unroll
  for (int v = 0; v < 8; ++v) {
    int m = mtile * 16 + v + ((lane >> 4) << 3);
    signed char out = 0;
    if (co < 252) out = csign(((float)c[v] + bias) * s + t0);
    h4[(size_t)m * 256 + co] = out;
  }
}

__global__ __launch_bounds__(32) void fc2_wmma_kernel(
    const signed char* __restrict__ h4, const signed char* __restrict__ fw2s,
    const float* __restrict__ fb2, const float* __restrict__ g, const float* __restrict__ be,
    const float* __restrict__ mm, const float* __restrict__ vv,
    float* __restrict__ logits) {
  int mtile = blockIdx.x;
  v8i c = {0, 0, 0, 0, 0, 0, 0, 0};
#pragma unroll
  for (int kb = 0; kb < 256; kb += 64) {
    v8i a = load_a_frag(h4, 256, mtile * 16, kb, 256);
    v8i b = load_b_frag(fw2s, 256, 0, kb);
    c = __builtin_amdgcn_wmma_i32_16x16x64_iu8(true, a, true, b, c, false, false);
  }
  int lane = threadIdx.x & 31;
  int co = lane & 15;
  if (co < 10) {
    float bias = fb2[co];
    float s = g[co] * rsqrtf(vv[co] + EPS);
    float t0 = be[co] - mm[co] * s;
#pragma unroll
    for (int v = 0; v < 8; ++v) {
      int m = mtile * 16 + v + ((lane >> 4) << 3);
      logits[(size_t)m * 10 + co] = ((float)c[v] + bias) * s + t0;
    }
  }
}

__global__ void logsoftmax_kernel(const float* __restrict__ logits, float* __restrict__ out) {
  int r = blockIdx.x * blockDim.x + threadIdx.x;
  if (r >= 4096) return;
  const float* z = logits + (size_t)r * 10;
  float mx = z[0];
#pragma unroll
  for (int i = 1; i < 10; ++i) mx = fmaxf(mx, z[i]);
  float sum = 0.f;
#pragma unroll
  for (int i = 0; i < 10; ++i) sum += __expf(z[i] - mx);
  float lse = mx + __logf(sum);
#pragma unroll
  for (int i = 0; i < 10; ++i) out[(size_t)r * 10 + i] = z[i] - lse;
}

extern "C" void kernel_launch(void* const* d_in, const int* in_sizes, int n_in,
                              void* d_out, int out_size, void* d_ws, size_t ws_size,
                              hipStream_t stream) {
  (void)in_sizes; (void)n_in; (void)out_size; (void)ws_size;
  const float* x   = (const float*)d_in[0];
  const float* w1  = (const float*)d_in[1];
  const float* cb1 = (const float*)d_in[2];
  const float* g1  = (const float*)d_in[3];
  const float* be1 = (const float*)d_in[4];
  const float* m1  = (const float*)d_in[5];
  const float* v1  = (const float*)d_in[6];
  const float* w2  = (const float*)d_in[7];
  const float* cb2 = (const float*)d_in[8];
  const float* g2  = (const float*)d_in[9];
  const float* be2 = (const float*)d_in[10];
  const float* m2  = (const float*)d_in[11];
  const float* v2  = (const float*)d_in[12];
  const float* w3  = (const float*)d_in[13];
  const float* cb3 = (const float*)d_in[14];
  const float* g3  = (const float*)d_in[15];
  const float* be3 = (const float*)d_in[16];
  const float* m3  = (const float*)d_in[17];
  const float* v3  = (const float*)d_in[18];
  const float* fw1 = (const float*)d_in[19];
  const float* fb1 = (const float*)d_in[20];
  const float* g4  = (const float*)d_in[21];
  const float* be4 = (const float*)d_in[22];
  const float* m4  = (const float*)d_in[23];
  const float* v4  = (const float*)d_in[24];
  const float* fw2 = (const float*)d_in[25];
  const float* fb2 = (const float*)d_in[26];
  const float* g5  = (const float*)d_in[27];
  const float* be5 = (const float*)d_in[28];
  const float* m5  = (const float*)d_in[29];
  const float* v5  = (const float*)d_in[30];
  float* out = (float*)d_out;

  char* ws = (char*)d_ws;
  signed char* a1   = (signed char*)(ws + 0);
  signed char* w2s  = (signed char*)(ws + 14450688);
  signed char* w3s  = (signed char*)(ws + 14475264);
  signed char* fw1s = (signed char*)(ws + 14621184);
  signed char* fw2s = (signed char*)(ws + 14653952);
  signed char* im2  = (signed char*)(ws + 14658048);
  signed char* a2   = (signed char*)(ws + 224373248);
  signed char* a3   = (signed char*)(ws + 229288448);
  signed char* h4   = (signed char*)(ws + 229812736);
  float*       lgt  = (float*)      (ws + 230861312);

  binpad_kernel<<<(48 * 512 + 255) / 256, 256, 0, stream>>>(w2, w2s, 48, 450, 48, 512);
  binpad_kernel<<<(120 * 1216 + 255) / 256, 256, 0, stream>>>(w3, w3s, 120, 1200, 120, 1216);
  binpad_kernel<<<(256 * 128 + 255) / 256, 256, 0, stream>>>(fw1, fw1s, 252, 120, 256, 128);
  binpad_kernel<<<(16 * 256 + 255) / 256, 256, 0, stream>>>(fw2, fw2s, 10, 252, 16, 256);

  {
    int total = 4096 * 18 * 196;
    conv1_kernel<<<(total + 255) / 256, 256, 0, stream>>>(x, w1, cb1, g1, be1, m1, v1, a1);
  }

  {
    int total = 409600 * 128;
    im2col2_kernel<<<(total + 255) / 256, 256, 0, stream>>>(a1, im2);
  }
  conv2_wmma_kernel<<<409600 / 16, 32, 0, stream>>>(im2, w2s, cb2, g2, be2, m2, v2, a2);

  conv3_wmma_kernel<<<dim3(4096 / 16, 8), 32, 0, stream>>>(a2, w3s, cb3, g3, be3, m3, v3, a3);
  fc1_wmma_kernel<<<dim3(4096 / 16, 16), 32, 0, stream>>>(a3, fw1s, fb1, g4, be4, m4, v4, h4);
  fc2_wmma_kernel<<<dim3(4096 / 16, 1), 32, 0, stream>>>(h4, fw2s, fb2, g5, be5, m5, v5, lgt);

  logsoftmax_kernel<<<(4096 + 255) / 256, 256, 0, stream>>>(lgt, out);
}
